// ODE_LSTM_10496900072113
// MI455X (gfx1250) — matrix operand order repaired and checked by reading, not compiled, not run on hardware
//
#include <hip/hip_runtime.h>

// ---------------- Problem constants ----------------
#define BB   256      // batch
#define TT   512      // time steps
#define DD   300      // input feature dim
#define KP   320      // D padded to multiple of 32 for WMMA K
#define NOUT 2048     // 4*S*U gate columns
#define SU   512      // S*U state size
#define UU   128      // units per scale

typedef __attribute__((ext_vector_type(16))) __bf16       v16bf;
typedef __attribute__((ext_vector_type(8)))  float        v8f;
typedef __attribute__((ext_vector_type(4)))  unsigned int v4u;
typedef __attribute__((ext_vector_type(4)))  unsigned int u32x4;
typedef __attribute__((ext_vector_type(8)))  int          i32x8;
typedef __attribute__((ext_vector_type(4)))  int          i32x4;

union FragBF {
    v16bf          v;
    v4u            u[2];
    unsigned short us[16];
};

__device__ __forceinline__ unsigned short f2bf_u(float f) {
    union { float f; unsigned u; } x; x.f = f;
    unsigned u = x.u;
    unsigned r = (u + 0x7fffu + ((u >> 16) & 1u)) >> 16;   // round-to-nearest-even
    return (unsigned short)r;
}

__device__ __forceinline__ float bf2f(unsigned short s) {
    union { unsigned u; float f; } x; x.u = ((unsigned)s) << 16;
    return x.f;
}

__device__ __forceinline__ float sigmoid_f(float x) {
    return 1.0f / (1.0f + __expf(-x));
}

// TDM: DMA a [16 x 2048] bf16 tile (row stride 2048 elems, contiguous) into LDS.
// D# per CDNA5 ISA ch.8: group0 = {flags, lds_addr, global_addr, type=2},
// group1 = {data_size=2B, tensor dims, tile 2048x16, stride 2048}.
__device__ __forceinline__ void tdm_load_xi(const unsigned short* gsrc, void* lds_dst) {
    unsigned long long ga = (unsigned long long)(size_t)gsrc;
    unsigned lo = (unsigned)(size_t)lds_dst;        // LDS byte offset (low 32 bits)
    u32x4 g0;
    g0[0] = 1u;                                     // count=1, user descriptor
    g0[1] = lo;                                     // lds_addr
    g0[2] = (unsigned)(ga & 0xFFFFFFFFull);         // global_addr[31:0]
    g0[3] = (unsigned)((ga >> 32) & 0x1FFFFFFu) | (2u << 30);   // addr[56:32] | type=2
    i32x8 g1;
    g1[0] = (1 << 16);                  // workgroup_mask=0, data_size=1 (2 bytes)
    g1[1] = (int)(2048u << 16);         // tensor_dim0[15:0] = 2048
    g1[2] = 0;                          // tensor_dim0[31:16]=0, tensor_dim1[15:0]=0
    g1[3] = (int)((0x800u << 16) | 2u); // tile_dim0=2048, tensor_dim1[31:16]=2 (dim1=131072)
    g1[4] = 16;                         // tile_dim1=16, tile_dim2=0
    g1[5] = 2048;                       // tensor_dim0_stride[31:0] = 2048
    g1[6] = 0;                          // stride0[47:32]=0, stride1[15:0]=0
    g1[7] = 0;
    i32x4 z4 = {0, 0, 0, 0};
    i32x8 z8 = {0, 0, 0, 0, 0, 0, 0, 0};
    __builtin_amdgcn_tensor_load_to_lds(g0, g1, z4, z4, z8, 0);
}

// ---------------- Conversion / transpose kernels ----------------

// A_bf16[t*BB + b][k] = (k < DD) ? inputs[b][t][k] : 0   (time-major rows)
__global__ void conv_inputs(const float* __restrict__ in, unsigned short* __restrict__ A,
                            int total) {
    int i = blockIdx.x * blockDim.x + threadIdx.x;
    if (i >= total) return;
    int k   = i % KP;
    int row = i / KP;
    int t   = row >> 8;        // row / 256
    int b   = row & 255;
    float v = (k < DD) ? in[((size_t)b * TT + t) * DD + k] : 0.0f;
    A[i] = f2bf_u(v);
}

// BT[n][k] = (k < DD) ? kernel[k][n] : 0   (pre-transposed B operand, [NOUT, KP])
__global__ void conv_kernelT(const float* __restrict__ w, unsigned short* __restrict__ BT,
                             int total) {
    int i = blockIdx.x * blockDim.x + threadIdx.x;
    if (i >= total) return;
    int k = i % KP;
    int n = i / KP;
    float v = (k < DD) ? w[(size_t)k * NOUT + n] : 0.0f;
    BT[i] = f2bf_u(v);
}

// rwT[col][u] = rw[s][g][u] with col = s*512 + g, rw = reshape(recurrent_kernel,(4,512,128))
__global__ void conv_rwT(const float* __restrict__ rk, unsigned short* __restrict__ RwT,
                         int total) {
    int i = blockIdx.x * blockDim.x + threadIdx.x;
    if (i >= total) return;
    int u   = i & 127;
    int col = i >> 7;
    int s   = col >> 9;
    int g   = col & 511;
    RwT[i] = f2bf_u(rk[(size_t)s * (512 * 128) + (size_t)g * 128 + u]);
}

// ---------------- Phase 1: input projection GEMM ----------------
// Xi[T*B, NOUT] = A_bf16[T*B, KP] @ kernel (+bias), bf16 WMMA, f32 accumulate,
// result stored bf16. Workgroup tile 128x128; 8 waves 4x2; wave tile 32x64.
__global__ __launch_bounds__(256) void gemm_input_proj(
        const unsigned short* __restrict__ A,    // [M, KP] bf16
        const unsigned short* __restrict__ BT,   // [NOUT, KP] bf16 (B transposed)
        const float* __restrict__ bias,          // [NOUT]
        unsigned short* __restrict__ Xi)         // [M, NOUT] bf16
{
    const int lane = threadIdx.x & 31;
    const int wv   = threadIdx.x >> 5;
    const int mw   = wv & 3;        // 4 waves along M
    const int nw   = wv >> 2;       // 2 waves along N
    const int m_base = blockIdx.x * 128 + mw * 32;
    const int n_base = blockIdx.y * 128 + nw * 64;
    const int r    = lane & 15;
    const int half = lane >> 4;

    v8f acc[2][4];
#pragma unroll
    for (int j = 0; j < 4; ++j) {
        float bv = bias[n_base + j * 16 + r];    // C-layout: N = lane%16
#pragma unroll
        for (int i = 0; i < 2; ++i) {
            v8f a;
#pragma unroll
            for (int e = 0; e < 8; ++e) a[e] = bv;
            acc[i][j] = a;
        }
    }

    for (int kk = 0; kk < KP; kk += 32) {
        FragBF af[2], bf[4];
#pragma unroll
        for (int i = 0; i < 2; ++i) {
            const unsigned short* p = A + (size_t)(m_base + i * 16 + r) * KP + kk + half * 8;
            af[i].u[0] = *(const v4u*)p;          // K = half*8 .. +7
            af[i].u[1] = *(const v4u*)(p + 16);   // K = 16 + half*8 .. +7
        }
#pragma unroll
        for (int j = 0; j < 4; ++j) {
            const unsigned short* p = BT + (size_t)(n_base + j * 16 + r) * KP + kk + half * 8;
            bf[j].u[0] = *(const v4u*)p;          // K = half*8 .. +7
            bf[j].u[1] = *(const v4u*)(p + 16);   // K = 16 + half*8 .. +7
        }
#pragma unroll
        for (int i = 0; i < 2; ++i)
#pragma unroll
            for (int j = 0; j < 4; ++j)
                acc[i][j] = __builtin_amdgcn_wmma_f32_16x16x32_bf16(
                    false, af[i].v, false, bf[j].v, (short)0, acc[i][j], false, false);
    }

#pragma unroll
    for (int i = 0; i < 2; ++i)
#pragma unroll
        for (int j = 0; j < 4; ++j) {
            int n = n_base + j * 16 + r;
#pragma unroll
            for (int e = 0; e < 8; ++e) {
                int m = m_base + i * 16 + e + half * 8;
                Xi[(size_t)m * NOUT + n] = f2bf_u(acc[i][j][e]);
            }
        }
}

// ---------------- Phase 2: recurrent scan ----------------
// One workgroup (16 waves / 512 threads) owns 16 batch rows for all T steps.
// Per step: TDM-staged x_ih slab (bf16, LDS) -> acc init; x_hh via WMMA
// (K=128 per gate-scale, recurrent weights L2-resident); fused gate math
// updates c,h in LDS. TDM for step t+1 overlaps step t compute.
__global__ __launch_bounds__(512) void lstm_scan(
        const unsigned short* __restrict__ Xi,   // [T*B, NOUT] bf16 (bias folded)
        const unsigned short* __restrict__ RwT,  // [NOUT, UU] bf16
        float* __restrict__ out)                 // [B, SU]
{
    extern __shared__ float smem[];
    float*          xo    = smem;                          // [16][2048] f32  128 KB
    float*          hs    = smem + 16 * NOUT;              // [16][512]  f32   32 KB
    float*          cs    = hs + 16 * SU;                  // [16][512]  f32   32 KB
    unsigned short* stage = (unsigned short*)(cs + 16 * SU); // [16][2048] bf16 64 KB

    const int tid  = threadIdx.x;
    const int lane = tid & 31;
    const int wv   = tid >> 5;        // 16 waves; wave w -> cols [w*128, w*128+128)
    const int b0   = blockIdx.x * 16;
    const int r    = lane & 15;
    const int half = lane >> 4;
    const int s    = wv >> 2;         // gate-scale for this wave's columns

    for (int i = tid; i < 16 * SU; i += 512) { hs[i] = 0.0f; cs[i] = 0.0f; }
    if (wv == 0) {
        tdm_load_xi(Xi + ((size_t)0 * BB + b0) * NOUT, stage);
        __builtin_amdgcn_s_wait_tensorcnt(0);
    }
    __syncthreads();

    for (int t = 0; t < TT; ++t) {
        // ---- consume staged x_ih slab into accumulators (8 col tiles / wave) ----
        v8f acc[8];
#pragma unroll
        for (int nt = 0; nt < 8; ++nt) {
            const int n = wv * 128 + nt * 16 + r;
#pragma unroll
            for (int e = 0; e < 8; ++e)
                acc[nt][e] = bf2f(stage[(size_t)(e + half * 8) * NOUT + n]);
        }
        __syncthreads();   // everyone consumed stage

        // ---- kick off DMA of next step's slab; overlaps WMMA + gates ----
        if (wv == 0 && (t + 1) < TT)
            tdm_load_xi(Xi + ((size_t)(t + 1) * BB + b0) * NOUT, stage);

        // ---- A fragments: h slice for this wave's scale, f32->bf16 from LDS ----
        FragBF af[4];
#pragma unroll
        for (int kk = 0; kk < 4; ++kk) {
            const int kb = s * UU + kk * 32;
            const float* hp = &hs[r * SU + kb + half * 8];
#pragma unroll
            for (int e = 0; e < 8; ++e) af[kk].us[e]     = f2bf_u(hp[e]);
#pragma unroll
            for (int e = 0; e < 8; ++e) af[kk].us[8 + e] = f2bf_u(hp[16 + e]);
        }

        // ---- recurrent WMMA: 8 column tiles of 16 per wave ----
#pragma unroll
        for (int nt = 0; nt < 8; ++nt) {
            const int n = wv * 128 + nt * 16 + r;
#pragma unroll
            for (int kk = 0; kk < 4; ++kk) {
                FragBF bf;
                const unsigned short* bp = RwT + (size_t)n * UU + kk * 32 + half * 8;
                bf.u[0] = *(const v4u*)bp;
                bf.u[1] = *(const v4u*)(bp + 16);
                acc[nt] = __builtin_amdgcn_wmma_f32_16x16x32_bf16(
                    false, af[kk].v, false, bf.v, (short)0, acc[nt], false, false);
            }
#pragma unroll
            for (int e = 0; e < 8; ++e)
                xo[(size_t)(e + half * 8) * NOUT + n] = acc[nt][e];
        }
        __syncthreads();

        // ---- fused gates + state update ----
        for (int i = tid; i < 16 * SU; i += 512) {
            int b = i >> 9;            // row 0..15
            int j = i & 511;           // unit
            const float* x = &xo[(size_t)b * NOUT];
            float fg = sigmoid_f(x[j]);
            float ig = sigmoid_f(x[SU + j]);
            float og = sigmoid_f(x[2 * SU + j]);
            float gg = tanhf(x[3 * SU + j]);
            float c  = fg * cs[i] + ig * gg;
            cs[i] = c;
            hs[i] = og * tanhf(c);
        }
        if (wv == 0) __builtin_amdgcn_s_wait_tensorcnt(0);
        __syncthreads();   // stage ready + hs updated for next step
    }

    for (int i = tid; i < 16 * SU; i += 512) {
        int b = i >> 9;
        int j = i & 511;
        out[(size_t)(b0 + b) * SU + j] = hs[i];
    }
}

// ---------------- Host launch ----------------
extern "C" void kernel_launch(void* const* d_in, const int* in_sizes, int n_in,
                              void* d_out, int out_size, void* d_ws, size_t ws_size,
                              hipStream_t stream) {
    const float* inputs = (const float*)d_in[0];   // [256,512,300]
    const float* kernel = (const float*)d_in[1];   // [300,2048]
    const float* rker   = (const float*)d_in[2];   // [128,2048]
    const float* bias   = (const float*)d_in[3];   // [2048]
    float* out          = (float*)d_out;           // [256,512]

    char* ws = (char*)d_ws;
    const size_t M = (size_t)TT * BB;              // 131072
    size_t offA  = 0;                              // bf16 [M, KP]
    size_t offBT = offA  + M * KP * 2;             // bf16 [NOUT, KP]
    size_t offRw = offBT + (size_t)NOUT * KP * 2;  // bf16 [NOUT, UU]
    size_t offXi = offRw + (size_t)NOUT * UU * 2;  // bf16 [M, NOUT]

    unsigned short* A_bf  = (unsigned short*)(ws + offA);
    unsigned short* BT_bf = (unsigned short*)(ws + offBT);
    unsigned short* Rw_bf = (unsigned short*)(ws + offRw);
    unsigned short* Xi    = (unsigned short*)(ws + offXi);

    {
        int total = (int)(M * KP);
        conv_inputs<<<(total + 255) / 256, 256, 0, stream>>>(inputs, A_bf, total);
    }
    {
        int total = NOUT * KP;
        conv_kernelT<<<(total + 255) / 256, 256, 0, stream>>>(kernel, BT_bf, total);
    }
    {
        int total = NOUT * UU;
        conv_rwT<<<(total + 255) / 256, 256, 0, stream>>>(rker, Rw_bf, total);
    }

    dim3 g1((unsigned)(M / 128), NOUT / 128, 1);
    gemm_input_proj<<<g1, 256, 0, stream>>>(A_bf, BT_bf, bias, Xi);

    // LDS: xo(128K) + hs(32K) + cs(32K) + stage(64K) = 256 KB
    size_t lds_bytes = (size_t)(16 * NOUT + 2 * 16 * SU) * sizeof(float)
                     + (size_t)(16 * NOUT) * sizeof(unsigned short);
    lstm_scan<<<BB / 16, 512, lds_bytes, stream>>>(Xi, Rw_bf, out);
}
